// Basic_MPNN_9698036154710
// MI455X (gfx1250) — matrix operand order repaired and checked by reading, not compiled, not run on hardware
//
#include <hip/hip_runtime.h>

typedef __attribute__((ext_vector_type(16))) _Float16 v16h;
typedef __attribute__((ext_vector_type(8)))  float    v8f;

static constexpr int Bsz = 4;
static constexpr int Nn  = 512;
static constexpr int Dd  = 128;   // D == MID == OUT == 128

static __device__ __forceinline__ unsigned pack2h(float x, float y) {
  union { _Float16 h[2]; unsigned u; } p;
  p.h[0] = (_Float16)x; p.h[1] = (_Float16)y;
  return p.u;
}

// ---------------------------------------------------------------------------
// Kernel 1: small precompute GEMMs (msg1, msg2, h1, msgg) — negligible cost.
// One block (128 threads) per output row; node row staged in LDS.
// ---------------------------------------------------------------------------
__global__ __launch_bounds__(128)
void mpnn_pre(const float* __restrict__ node, const float* __restrict__ graph,
              const float* __restrict__ W1,  const float* __restrict__ b1,
              const float* __restrict__ W2,  const float* __restrict__ b2,
              const float* __restrict__ Wo1, const float* __restrict__ bo1,
              const float* __restrict__ Wg,  const float* __restrict__ bg,
              float* __restrict__ msg1, float* __restrict__ msg2,
              float* __restrict__ h1,   float* __restrict__ msgg)
{
  __shared__ float x[128];
  const int tid = threadIdx.x;
  const int bid = blockIdx.x;
  if (bid < 3 * Bsz * Nn) {
    const int mat = bid / (Bsz * Nn);
    const int row = bid % (Bsz * Nn);
    const float* W; const float* bias; float* dst;
    if (mat == 0)      { W = W1;  bias = b1;  dst = msg1; }
    else if (mat == 1) { W = W2;  bias = b2;  dst = msg2; }
    else               { W = Wo1; bias = bo1; dst = h1;   }
    x[tid] = node[(size_t)row * 128 + tid];
    __syncthreads();
    float acc = bias[tid];
#pragma unroll 4
    for (int k = 0; k < 128; ++k) acc = fmaf(x[k], W[k * 128 + tid], acc);
    dst[(size_t)row * 128 + tid] = acc;
  } else {
    const int bb = bid - 3 * Bsz * Nn;
    x[tid] = graph[bb * 128 + tid];
    __syncthreads();
    float acc = bg[tid];
#pragma unroll 4
    for (int k = 0; k < 128; ++k) acc = fmaf(x[k], Wg[k * 128 + tid], acc);
    msgg[bb * 128 + tid] = acc;
  }
}

// ---------------------------------------------------------------------------
// Kernel 2: fused edge-GEMM (WMMA f16->f32) + msg2 add + adjacency-masked max
// over senders + constant adds + Wo2 GEMV + relu.  One block per (b, i).
// B (We) fragments live in registers; 32 senders are staged per iteration and
// computed as TWO independent WMMA accumulator chains to hide WMMA->WMMA
// hazard latency.
// ---------------------------------------------------------------------------
__global__ __launch_bounds__(256)
void mpnn_fused(const float* __restrict__ edge,
                const int*   __restrict__ adj,
                const float* __restrict__ We,
                const float* __restrict__ be,
                const float* __restrict__ Wo2,
                const float* __restrict__ bo2,
                const float* __restrict__ msg1,
                const float* __restrict__ msg2,
                const float* __restrict__ h1,
                const float* __restrict__ msgg,
                float* __restrict__ out)
{
  __shared__ unsigned sWt[128 * 64];  // We transposed, f16-packed: sWt[m*64+k/2]
  __shared__ unsigned sA [32 * 64];   // edge tile (32 j rows), f16-packed
  __shared__ float    sM2[32 * 128];  // msg2 chunk (32 j rows)
  __shared__ int      sAdj[32];
  __shared__ float    sMsg[128];

  const int tid  = threadIdx.x;
  const int lane = tid & 31;
  const int wave = tid >> 5;
  const int bi   = blockIdx.x;
  const int b    = bi >> 9;          // / Nn
  const int i    = bi & (Nn - 1);

  // Stage We (k-major in memory) transposed+packed into LDS once per block.
  for (int idx = tid; idx < 128 * 64; idx += 256) {
    const int m  = idx >> 6;
    const int k2 = idx & 63;
    sWt[idx] = pack2h(We[(2 * k2) * 128 + m], We[(2 * k2 + 1) * 128 + m]);
  }
  __syncthreads();

  const int n    = lane & 15;        // N index within C tile / A row / B col
  const int half = lane >> 4;
  const int m    = (wave << 4) + n;  // this wave's m-column

  // Hoist loop-invariant B fragments (this wave's 16-wide m-tile of We) into
  // registers: 4 K-chunks x 8 VGPRs.
  // B 32x16 f16 layout: lane N=n; VGPR vv -> K = 2vv (+8 if vv>=4) (+8 upper half).
  union { v16h v; unsigned u[8]; } Bf[4];
#pragma unroll
  for (int kk = 0; kk < 4; ++kk) {
#pragma unroll
    for (int vv = 0; vv < 8; ++vv) {
      const int kbm = (kk << 5) + 2 * vv + ((vv >= 4) ? 8 : 0) + (half << 3);
      Bf[kk].u[vv] = sWt[m * 64 + (kbm >> 1)];
    }
  }

  float rmax[8];
#pragma unroll
  for (int p = 0; p < 8; ++p) rmax[p] = -3.0e38f;

  const size_t jstride = (size_t)Nn * Dd;                       // edge j-stride
  const size_t ebase   = (size_t)b * Nn * jstride + (size_t)i * Dd;

  for (int jc = 0; jc < 16; ++jc) {
    const int j0 = jc << 5;                      // 32 senders per iteration
    __syncthreads();   // previous iteration's consumers done before overwrite

    // Stage edge tile [32 j x 128 d] as packed f16 (8 float2 loads/thread).
    for (int idx = tid; idx < 32 * 64; idx += 256) {
      const int r  = idx >> 6;
      const int d2 = idx & 63;
      const float2 e = *(const float2*)(edge + ebase + (size_t)(j0 + r) * jstride + 2 * d2);
      sA[idx] = pack2h(e.x, e.y);
    }
    // Stage msg2 chunk [32 j x 128 m] (fp32).
    for (int idx = tid; idx < 32 * 128; idx += 256) {
      const int r  = idx >> 7;
      const int mm = idx & 127;
      sM2[idx] = msg2[(((size_t)b * Nn + (j0 + r)) << 7) + mm];
    }
    if (tid < 32) sAdj[tid] = adj[((size_t)b * Nn + (j0 + tid)) * Nn + i];
    __syncthreads();

    // Prefetch next iteration's edge rows (128B segments).
    if (jc + 1 < 16 && tid < 128) {
      const int r = tid >> 2, seg = tid & 3;
      __builtin_prefetch((const void*)(edge + ebase + (size_t)(j0 + 32 + r) * jstride + seg * 32), 0, 0);
    }

    // Two independent accumulator chains: rows j0..j0+15 (cA), j0+16..j0+31 (cB).
    v8f cA = {0.f, 0.f, 0.f, 0.f, 0.f, 0.f, 0.f, 0.f};
    v8f cB = {0.f, 0.f, 0.f, 0.f, 0.f, 0.f, 0.f, 0.f};
#pragma unroll
    for (int kk = 0; kk < 4; ++kk) {
      const int kb = kk << 5;
      union { v16h v; unsigned u[8]; } AfA, AfB;
#pragma unroll
      for (int vv = 0; vv < 8; ++vv) {
        // A 16x32 f16 layout: lane M=n; VGPR vv -> K = 2vv (+8 if vv>=4) (+8 upper half)
        const int ka = kb + 2 * vv + ((vv >= 4) ? 8 : 0) + (half << 3);
        AfA.u[vv] = sA[n * 64        + (ka >> 1)];
        AfB.u[vv] = sA[(16 + n) * 64 + (ka >> 1)];
      }
      cA = __builtin_amdgcn_wmma_f32_16x16x32_f16(false, AfA.v, false, Bf[kk].v,
                                                  (short)0, cA, false, false);
      cB = __builtin_amdgcn_wmma_f32_16x16x32_f16(false, AfB.v, false, Bf[kk].v,
                                                  (short)0, cB, false, false);
    }

    // C layout: VGPR p, lanes 0-15 -> M=p, lanes 16-31 -> M=p+8; N = lane%16.
#pragma unroll
    for (int p = 0; p < 8; ++p) {
      const int r = p + (half << 3);               // local sender row (chunk A)
      float va = cA[p] + sM2[r * 128 + m];
      va = sAdj[r] ? va : -3.0e38f;
      rmax[p] = fmaxf(rmax[p], va);
      float vb = cB[p] + sM2[(16 + r) * 128 + m];  // chunk B rows
      vb = sAdj[16 + r] ? vb : -3.0e38f;
      rmax[p] = fmaxf(rmax[p], vb);
    }
  }

  // Reduce over the 8 M-positions, then across lane halves (M vs M+8 sets).
  float cm = rmax[0];
#pragma unroll
  for (int p = 1; p < 8; ++p) cm = fmaxf(cm, rmax[p]);
  cm = fmaxf(cm, __shfl_xor(cm, 16, 32));

  const size_t rowo = ((size_t)b * Nn + i) << 7;
  if (lane < 16) {
    const float inner = cm;
    const float fin = (inner > -1.0e37f)
        ? inner + be[m] + msg1[rowo + m] + msgg[(b << 7) + m]
        : -1000000.0f;                 // no connected sender -> -BIG_NUMBER
    sMsg[m] = fin;
  }
  __syncthreads();

  // Final: h2 = msgs @ Wo2 + bo2 ; out = relu(h1 + h2)
  if (tid < 128) {
    float acc = bo2[tid];
#pragma unroll 4
    for (int k = 0; k < 128; ++k) acc = fmaf(sMsg[k], Wo2[k * 128 + tid], acc);
    const float h = h1[rowo + tid] + acc;
    out[rowo + tid] = fmaxf(h, 0.f);
  }
}

// ---------------------------------------------------------------------------
extern "C" void kernel_launch(void* const* d_in, const int* in_sizes, int n_in,
                              void* d_out, int out_size, void* d_ws, size_t ws_size,
                              hipStream_t stream) {
  const float* node  = (const float*)d_in[0];
  const float* edge  = (const float*)d_in[1];
  const float* graph = (const float*)d_in[2];
  const int*   adj   = (const int*)  d_in[3];
  // d_in[4] = hidden (unused by the reference)
  const float* W1  = (const float*)d_in[5];  const float* b1  = (const float*)d_in[6];
  const float* W2  = (const float*)d_in[7];  const float* b2  = (const float*)d_in[8];
  const float* We  = (const float*)d_in[9];  const float* be  = (const float*)d_in[10];
  const float* Wg  = (const float*)d_in[11]; const float* bg  = (const float*)d_in[12];
  const float* Wo1 = (const float*)d_in[13]; const float* bo1 = (const float*)d_in[14];
  const float* Wo2 = (const float*)d_in[15]; const float* bo2 = (const float*)d_in[16];
  float* out = (float*)d_out;

  float* ws   = (float*)d_ws;
  float* msg1 = ws;                      // [B,N,128]
  float* msg2 = ws + 262144;             // [B,N,128]
  float* h1   = ws + 524288;             // [B,N,128]
  float* msgg = ws + 786432;             // [B,128]

  mpnn_pre<<<3 * Bsz * Nn + Bsz, 128, 0, stream>>>(
      node, graph, W1, b1, W2, b2, Wo1, bo1, Wg, bg, msg1, msg2, h1, msgg);

  mpnn_fused<<<Bsz * Nn, 256, 0, stream>>>(
      edge, adj, We, be, Wo2, bo2, msg1, msg2, h1, msgg, out);
}
